// RNN_LSTM_EMBED_58248346468904
// MI455X (gfx1250) — matrix operand order repaired and checked by reading, not compiled, not run on hardware
//
#include <hip/hip_runtime.h>

// ---------------------------------------------------------------------------
// Problem constants (from reference): B=32, T=128, E=512, H=1024, V=32000
// ---------------------------------------------------------------------------
#define BB 32
#define TT 128
#define EE 512
#define HH 1024
#define G4H 4096
#define VV 32000

typedef __bf16  v16bf __attribute__((ext_vector_type(16)));
typedef float   v8f   __attribute__((ext_vector_type(8)));

union FragU { v16bf v; uint4 q[2]; };

__device__ __forceinline__ unsigned short f32_to_bf16(float f) {
    unsigned int u = __float_as_uint(f);
    unsigned int lsb = (u >> 16) & 1u;
    u += 0x7fffu + lsb;                 // round-to-nearest-even
    return (unsigned short)(u >> 16);
}

// A-fragment (16x32 bf16, row-major source, row stride lda elements):
// lane l: m = m0 + (l&15), half h = l>>4; elements 0..15 -> K = 16h..16h+15.
// Two global_load_b128 per lane.
__device__ __forceinline__ v16bf load_a_frag(const unsigned short* __restrict__ A,
                                             int lda, int m0, int lane) {
    int h = lane >> 4, m = m0 + (lane & 15);
    const unsigned short* p = A + (size_t)m * (size_t)lda;
    FragU f;
    f.q[0] = *(const uint4*)(p + 16 * h);
    f.q[1] = *(const uint4*)(p + 16 * h + 8);
    return f.v;
}

// B-fragment (32x16 bf16 = rows of a row-major [N,K] matrix):
// lane l: n = n0 + (l&15), half h = l>>4; elements 0..15 -> K = 16h..16h+15.
__device__ __forceinline__ v16bf load_b_frag(const unsigned short* __restrict__ B,
                                             int ldb, int n0, int lane) {
    int h = lane >> 4, n = n0 + (lane & 15);
    const unsigned short* p = B + (size_t)n * (size_t)ldb + 16 * h;
    FragU f;
    f.q[0] = *(const uint4*)(p);
    f.q[1] = *(const uint4*)(p + 8);
    return f.v;
}

__device__ __forceinline__ v8f wmma_bf16(v16bf a, v16bf b, v8f c) {
    // D(f32 16x16) = A(16x32 bf16) * B(32x16 bf16) + C
    return __builtin_amdgcn_wmma_f32_16x16x32_bf16(
        false, a, false, b, (short)0, c, false, false);
}

// ---------------------------------------------------------------------------
// f32 -> bf16 convert
// ---------------------------------------------------------------------------
__global__ void __launch_bounds__(256)
cvt_bf16_kernel(const float* __restrict__ in, unsigned short* __restrict__ out, int n) {
    int i = blockIdx.x * blockDim.x + threadIdx.x;
    if (i < n) out[i] = f32_to_bf16(in[i]);
}

// bsum = b_ih + b_hh
__global__ void __launch_bounds__(256)
bias_sum_kernel(const float* __restrict__ a, const float* __restrict__ b,
                float* __restrict__ o, int n) {
    int i = blockIdx.x * blockDim.x + threadIdx.x;
    if (i < n) o[i] = a[i] + b[i];
}

// x_emb_bf16[r*E + e] = bf16(embed[x[r]*E + e]),  r = b*T + t
__global__ void __launch_bounds__(256)
embed_gather_kernel(const int* __restrict__ x, const float* __restrict__ embed,
                    unsigned short* __restrict__ xe, int n) {
    int i = blockIdx.x * blockDim.x + threadIdx.x;
    if (i >= n) return;
    int e = i & (EE - 1);
    int r = i >> 9;                         // E = 512 = 2^9
    int idx = x[r];
    xe[i] = f32_to_bf16(embed[(size_t)idx * EE + e]);
}

// ---------------------------------------------------------------------------
// GEMM: D[M,N] = A[M,K] @ B[N,K]^T + bias[N]
// Per wave: 64(M) x 64(N) tile (16 WMMA accumulators, 4 A-frags x 4 B-frags
// per K-step of 32). Block = 4 waves in a 2x2 arrangement -> block tile
// 128x128. grid = (N/128, M/128), blockDim = 128.
// ---------------------------------------------------------------------------
__global__ void __launch_bounds__(128)
gemm_bf16_abt_kernel(const unsigned short* __restrict__ A, int lda,
                     const unsigned short* __restrict__ B, int ldb,
                     const float* __restrict__ bias,
                     float* __restrict__ D, int ldd, int K) {
    const int lane = threadIdx.x & 31;
    const int wave = threadIdx.x >> 5;
    const int m0 = (blockIdx.y * 2 + (wave >> 1)) * 64;
    const int n0 = (blockIdx.x * 2 + (wave & 1)) * 64;

    v8f acc[4][4] = {};
    for (int k = 0; k < K; k += 32) {
        v16bf a[4], b[4];
#pragma unroll
        for (int i = 0; i < 4; ++i) a[i] = load_a_frag(A + k, lda, m0 + 16 * i, lane);
#pragma unroll
        for (int j = 0; j < 4; ++j) b[j] = load_b_frag(B + k, ldb, n0 + 16 * j, lane);
#pragma unroll
        for (int i = 0; i < 4; ++i)
#pragma unroll
            for (int j = 0; j < 4; ++j)
                acc[i][j] = wmma_bf16(a[i], b[j], acc[i][j]);
    }

    const int n = lane & 15;
    const int h = lane >> 4;
    float bv[4];
#pragma unroll
    for (int j = 0; j < 4; ++j) bv[j] = bias[n0 + 16 * j + n];
#pragma unroll
    for (int i = 0; i < 4; ++i) {
#pragma unroll
        for (int e = 0; e < 8; ++e) {
            int row = m0 + 16 * i + e + 8 * h;   // C/D layout: VGPR e -> M = e + 8*half
            float* dr = D + (size_t)row * ldd;
#pragma unroll
            for (int j = 0; j < 4; ++j)
                dr[n0 + 16 * j + n] = acc[i][j][e] + bv[j];
        }
    }
}

// ---------------------------------------------------------------------------
// One LSTM timestep, fully fused, with K split across the 4 waves of a block:
//   wave w accumulates gates over K in [w*256, w*256+256) for its block's
//   16 hidden columns (all 4 gates, all 32 batch rows -> 8 WMMA accs/wave),
//   partial sums are reduced through LDS, then all 128 threads apply the
//   i/f/g/o nonlinearities and the c/h update.
// grid = H/16 = 64 blocks, blockDim = 128 (4 waves).
// ---------------------------------------------------------------------------
__global__ void __launch_bounds__(128)
lstm_step_kernel(const unsigned short* __restrict__ hs,   // [B*T, H] bf16 history
                 const unsigned short* __restrict__ Whh,  // [4H, H] bf16
                 const float* __restrict__ xg,            // [B*T, 4H] f32
                 float* __restrict__ c,                   // [B, H] f32 state
                 unsigned short* __restrict__ hs_out,     // == hs (write slot t)
                 int t) {
    const int lane = threadIdx.x & 31;
    const int wave = threadIdx.x >> 5;
    const int j0 = blockIdx.x * 16;                 // this block's hidden columns

    // partial accumulators: red[wave][gate][p][lane][e]
    __shared__ float red[4 * 4 * 2 * 32 * 8];       // 32 KB

    v8f acc[4][2] = {};
    if (t > 0) {
        const unsigned short* hPrev = hs + (size_t)(t - 1) * HH;  // row b at b*T*H
        const int rsH = TT * HH;                                   // batch row stride
        const int kBeg = wave * (HH / 4);
        const int kEnd = kBeg + (HH / 4);
        for (int k = kBeg; k < kEnd; k += 32) {
            v16bf a0 = load_a_frag(hPrev + k, rsH, 0,  lane);
            v16bf a1 = load_a_frag(hPrev + k, rsH, 16, lane);
#pragma unroll
            for (int g = 0; g < 4; ++g) {
                v16bf b = load_b_frag(Whh + (size_t)g * HH * HH + k, HH, j0, lane);
                acc[g][0] = wmma_bf16(a0, b, acc[g][0]);
                acc[g][1] = wmma_bf16(a1, b, acc[g][1]);
            }
        }
    }

    // stage partial sums to LDS (two ds_store_b128 per acc)
#pragma unroll
    for (int g = 0; g < 4; ++g)
#pragma unroll
        for (int p = 0; p < 2; ++p) {
            int off = (((wave * 4 + g) * 2 + p) * 32 + lane) * 8;
            *(v8f*)&red[off] = acc[g][p];
        }
    __syncthreads();

    // pointwise LSTM update: 512 (b, col) elements, 4 per thread
    const int tid = threadIdx.x;
    for (int e = tid; e < BB * 16; e += 128) {
        const int b  = e >> 4;                       // batch row
        const int ci = e & 15;                       // column within block tile
        const int col = j0 + ci;
        // invert fragment mapping: b = p*16 + j + 8*hh, lane = ci + 16*hh
        const int p  = b >> 4;
        const int jj = b & 7;
        const int hh = (b >> 3) & 1;
        const int ls = ci + 16 * hh;

        const float* xr = xg + ((size_t)b * TT + t) * G4H;
        float gate[4];
#pragma unroll
        for (int g = 0; g < 4; ++g) {
            float s = xr[g * HH + col];
#pragma unroll
            for (int w = 0; w < 4; ++w)
                s += red[(((w * 4 + g) * 2 + p) * 32 + ls) * 8 + jj];
            gate[g] = s;
        }
        float iv = 1.0f / (1.0f + __expf(-gate[0]));
        float fv = 1.0f / (1.0f + __expf(-gate[1]));
        float gv = tanhf(gate[2]);
        float ov = 1.0f / (1.0f + __expf(-gate[3]));
        float cold = (t > 0) ? c[(size_t)b * HH + col] : 0.0f;
        float cnew = fv * cold + iv * gv;
        c[(size_t)b * HH + col] = cnew;
        float hv = ov * tanhf(cnew);
        hs_out[((size_t)b * TT + t) * HH + col] = f32_to_bf16(hv);
    }
}

// ---------------------------------------------------------------------------
// Host-side launch
// ---------------------------------------------------------------------------
extern "C" void kernel_launch(void* const* d_in, const int* in_sizes, int n_in,
                              void* d_out, int out_size, void* d_ws, size_t ws_size,
                              hipStream_t stream) {
    const int*   x     = (const int*)  d_in[0];   // [B,T]
    const float* embed = (const float*)d_in[1];   // [V,E]
    const float* W_ih  = (const float*)d_in[2];   // [4H,E]
    const float* W_hh  = (const float*)d_in[3];   // [4H,H]
    const float* b_ih  = (const float*)d_in[4];   // [4H]
    const float* b_hh  = (const float*)d_in[5];   // [4H]
    const float* fc_w  = (const float*)d_in[6];   // [V,H]
    const float* fc_b  = (const float*)d_in[7];   // [V]
    float* out = (float*)d_out;                   // [B,T,V]

    char* ws = (char*)d_ws;
    // Workspace layout (all offsets 256B-aligned)
    unsigned short* W_ih_b = (unsigned short*)(ws + 0);           //  4 MB: 4096*512
    unsigned short* W_hh_b = (unsigned short*)(ws + 4194304);     //  8 MB: 4096*1024
    unsigned short* fc_w_b = (unsigned short*)(ws + 12582912);    // 64 MB: 32000*1024
    unsigned short* xe     = (unsigned short*)(ws + 78118912);    //  4 MB: 4096*512
    float*          xg     = (float*)         (ws + 82313216);    // 64 MB: 4096*4096
    float*          bsum   = (float*)         (ws + 149422080);   // 16 KB: 4096
    float*          cbuf   = (float*)         (ws + 149438464);   // 128 KB: 32*1024
    unsigned short* hsbuf  = (unsigned short*)(ws + 149569536);   //  8 MB: 4096*1024

    // 1) weight converts + bias sum
    cvt_bf16_kernel<<<(G4H * EE) / 256, 256, 0, stream>>>(W_ih, W_ih_b, G4H * EE);
    cvt_bf16_kernel<<<(G4H * HH) / 256, 256, 0, stream>>>(W_hh, W_hh_b, G4H * HH);
    cvt_bf16_kernel<<<(VV * HH) / 256, 256, 0, stream>>>(fc_w, fc_w_b, VV * HH);
    bias_sum_kernel<<<G4H / 256, 256, 0, stream>>>(b_ih, b_hh, bsum, G4H);

    // 2) embedding gather -> bf16
    embed_gather_kernel<<<(BB * TT * EE) / 256, 256, 0, stream>>>(x, embed, xe, BB * TT * EE);

    // 3) input projection: xg[4096,4096] = xe @ W_ih^T + (b_ih + b_hh)
    {
        dim3 grid(G4H / 128, (BB * TT) / 128);
        gemm_bf16_abt_kernel<<<grid, 128, 0, stream>>>(xe, EE, W_ih_b, EE, bsum,
                                                       xg, G4H, EE);
    }

    // 4) sequential LSTM scan: one fused kernel per timestep
    for (int t = 0; t < TT; ++t) {
        lstm_step_kernel<<<HH / 16, 128, 0, stream>>>(hsbuf, W_hh_b, xg, cbuf, hsbuf, t);
    }

    // 5) classifier: out[4096,32000] = hs @ fc_w^T + fc_b
    {
        dim3 grid(VV / 128, (BB * TT) / 128);
        gemm_bf16_abt_kernel<<<grid, 128, 0, stream>>>(hsbuf, HH, fc_w_b, HH, fc_b,
                                                       out, VV, HH);
    }
}
